// Decoder_26620207301229
// MI455X (gfx1250) — compile-verified, matrix operand order checked by reading, not run on hardware
//
#include <hip/hip_runtime.h>
#include <cstdint>

#define N_NODES 20000
#define N_EDGES 640000
#define LATENT 512
#define HID 256
#define OUT_C 128
#define HEADS 2
#define NEG_SLOPE 0.2f

typedef __bf16 bf16_t;
typedef bf16_t v16bf __attribute__((ext_vector_type(16)));
typedef float v8f __attribute__((ext_vector_type(8)));
typedef unsigned int u32x4 __attribute__((ext_vector_type(4)));
typedef int i32x4 __attribute__((ext_vector_type(4)));
typedef int i32x8 __attribute__((ext_vector_type(8)));
typedef __attribute__((address_space(3))) unsigned short lds_ushort;

union FragBf { v16bf v; uint4 q[2]; };

// native RNE f32->bf16 via hardware cvt (v_cvt_pk_bf16_f32 / v_cvt_bf16_f32)
__device__ __forceinline__ unsigned pk2bf(float a, float b) {
  union { unsigned u; __bf16 h[2]; } p;
  p.h[0] = (__bf16)a;
  p.h[1] = (__bf16)b;
  return p.u;
}

// ---------------------------------------------------------------------------
// Weight transpose + f32 -> bf16 :  Wt[n*K + k] = bf16(W[k*N + n])
// ---------------------------------------------------------------------------
__global__ void k_wt_bf16(const float* __restrict__ W,
                          unsigned short* __restrict__ Wt, int K, int Nf) {
  int idx = blockIdx.x * blockDim.x + threadIdx.x;
  if (idx >= K * Nf) return;
  int k = idx / Nf, n = idx % Nf;
  union { unsigned short u; __bf16 h; } p;
  p.h = (__bf16)W[idx];
  Wt[(size_t)n * K + k] = p.u;
}

// ---------------------------------------------------------------------------
// TDM: async-load a 128x32 bf16 tile (row-major, row stride K elems) from
// global into LDS with hardware padding: +4 DWORDs every 16 DWORDs, i.e.
// 64B data + 16B pad per row -> LDS row stride of 40 ushorts (conflict-free
// for the WMMA fragment loads). 2D tile -> descriptor groups 0 and 1 only.
// ---------------------------------------------------------------------------
__device__ __forceinline__ void tdm_load_b_tile(unsigned lds_off,
                                                const unsigned short* gptr,
                                                int K) {
  unsigned long long ga = (unsigned long long)gptr;
  u32x4 g0;
  g0[0] = 1u;                                            // count=1 (user D#)
  g0[1] = lds_off;                                       // lds_addr (bytes)
  g0[2] = (unsigned)(ga & 0xFFFFFFFFull);                // global_addr[31:0]
  g0[3] = (unsigned)((ga >> 32) & 0x1FFFFFFull)          // global_addr[56:32]
        | (2u << 30);                                    // type=2 ("image")
  i32x8 g1;
  g1[0] = (int)((1u << 16)      // data_size = 2 bytes
              | (1u << 20)      // pad_enable
              | (3u << 22)      // pad_interval: 16 DWORDs (64B) between pads
              | (3u << 25));    // pad_amount: 4 DWORDs (16B)
  g1[1] = (int)(32u << 16);     // tensor_dim0[15:0] = 32
  g1[2] = (int)(128u << 16);    // tensor_dim0 hi = 0 | tensor_dim1[15:0] = 128
  g1[3] = (int)(32u << 16);     // tensor_dim1 hi = 0 | tile_dim0 = 32
  g1[4] = (int)128;             // tile_dim1 = 128, tile_dim2 = 0
  g1[5] = (int)(unsigned)K;     // tensor_dim0_stride[31:0] = K
  g1[6] = 0;                    // stride hi | tensor_dim1_stride lo (unused, 2D)
  g1[7] = 0;
  i32x4 z4 = {0, 0, 0, 0};
#if __has_include(<hip/amd_detail/amd_gfx1250_TDM.h>)
  i32x8 z8 = {0, 0, 0, 0, 0, 0, 0, 0};
  __builtin_amdgcn_tensor_load_to_lds(g0, g1, z4, z4, z8, 0);
#else
  __builtin_amdgcn_tensor_load_to_lds(g0, g1, z4, z4, 0);
#endif
}

// ---------------------------------------------------------------------------
// Tiled bf16 WMMA GEMM:  C = act(A[M,K] @ B[K,Nf] + bias)
//   A: f32 row-major (cvt to bf16 while staging);  Bt: bf16 [Nf][K] via TDM.
//   Block = 256 threads = 8 waves (2 x 4), tile 64(M) x 128(N), K-step 32.
//   Software-pipelined: double-buffered LDS, TDM(i+1) + stageA(i+1) overlap
//   the WMMAs of step i; one barrier per K-step.
// ---------------------------------------------------------------------------
__global__ __launch_bounds__(256) void k_gemm(const float* __restrict__ A,
                                              const unsigned short* __restrict__ Bt,
                                              const float* __restrict__ bias,
                                              float* __restrict__ C,
                                              int M, int K, int Nf, int relu) {
  __shared__ unsigned short lA[2][64 * 40];    // 64 rows, K-strip 32, pad->40
  __shared__ unsigned short lB[2][128 * 40];   // 128 cols(N), K-strip 32, pad->40

  const int t    = threadIdx.x;
  const int lane = t & 31;
  const int wave = t >> 5;
  const int wm   = wave & 1;    // M sub-offset wm*32
  const int wn   = wave >> 1;   // N sub-offset wn*32
  const int row0 = blockIdx.x * 64;
  const int col0 = blockIdx.y * 128;

  unsigned lBoff[2];
  lBoff[0] = (unsigned)(unsigned long long)(lds_ushort*)&lB[0][0];
  lBoff[1] = (unsigned)(unsigned long long)(lds_ushort*)&lB[1][0];
  const unsigned short* btBase = Bt + (size_t)col0 * K;

  // A staging coords (per thread): 64x32 tile, 8 elems/thread
  const int ar = t >> 2;
  const int ac = (t & 3) * 8;
  int gr = row0 + ar; if (gr > M - 1) gr = M - 1;        // clamp ragged M
  const float* aRow = A + (size_t)gr * K + ac;

  auto stageA = [&](int k0, int buf) {
    const float4* ap = (const float4*)(aRow + k0);
    float4 f0 = ap[0];
    float4 f1 = ap[1];
    unsigned int* d = (unsigned int*)&lA[buf][ar * 40 + ac];
    d[0] = pk2bf(f0.x, f0.y);
    d[1] = pk2bf(f0.z, f0.w);
    d[2] = pk2bf(f1.x, f1.y);
    d[3] = pk2bf(f1.z, f1.w);
  };

  v8f acc[2][2] = {};

  const int lrow = lane & 15;
  const int kh   = (lane >> 4) * 8;   // hi-half lanes start at K=8 / K=24

  const int steps = K >> 5;

  // ---- prologue: fill buffer 0
  if (wave == 0) tdm_load_b_tile(lBoff[0], btBase, K);
  stageA(0, 0);
  if (wave == 0) __builtin_amdgcn_s_wait_tensorcnt(0);
  __syncthreads();

  for (int i = 0; i < steps; ++i) {
    const int cur = i & 1;
    const int nxt = cur ^ 1;
    const bool more = (i + 1 < steps);

    // ---- prefetch next K-strip into the other buffer (overlaps WMMAs)
    if (more) {
      if (wave == 0) tdm_load_b_tile(lBoff[nxt], btBase + (i + 1) * 32, K);
      stageA((i + 1) * 32, nxt);
    }

    // ---- fragment loads (16-bit A layout: lane=row/col, kh by lane half)
    FragBf fa[2], fb[2];
#pragma unroll
    for (int ii = 0; ii < 2; ++ii) {
      const unsigned short* base = &lA[cur][(wm * 32 + ii * 16 + lrow) * 40 + kh];
      fa[ii].q[0] = *(const uint4*)(base);
      fa[ii].q[1] = *(const uint4*)(base + 16);
    }
#pragma unroll
    for (int jj = 0; jj < 2; ++jj) {
      const unsigned short* base = &lB[cur][(wn * 32 + jj * 16 + lrow) * 40 + kh];
      fb[jj].q[0] = *(const uint4*)(base);
      fb[jj].q[1] = *(const uint4*)(base + 16);
    }

#pragma unroll
    for (int ii = 0; ii < 2; ++ii)
#pragma unroll
      for (int jj = 0; jj < 2; ++jj)
        acc[ii][jj] = __builtin_amdgcn_wmma_f32_16x16x32_bf16(
            false, fa[ii].v, false, fb[jj].v, (short)0, acc[ii][jj], false, false);

    if (more) {
      if (wave == 0) __builtin_amdgcn_s_wait_tensorcnt(0);
      __syncthreads();   // publish nxt buffers; retires all reads of cur
    }
  }

  // ---- epilogue: C/D layout: VGPR r -> M = r + 8*(lane>=16), N = lane&15
  const size_t ldc = (size_t)Nf;
  const bool full = (row0 + 64 <= M);   // uniform per block
#pragma unroll
  for (int i = 0; i < 2; ++i) {
    int mbase = row0 + wm * 32 + i * 16 + (lane >> 4) * 8;
#pragma unroll
    for (int j = 0; j < 2; ++j) {
      int n = col0 + wn * 32 + j * 16 + (lane & 15);
      float bv = bias ? bias[n] : 0.0f;
      float* cp = C + (size_t)mbase * ldc + n;
      if (full) {
#pragma unroll
        for (int r = 0; r < 8; ++r) {
          float v = acc[i][j][r] + bv;
          if (relu) v = fmaxf(v, 0.0f);
          cp[(size_t)r * ldc] = v;
        }
      } else {
#pragma unroll
        for (int r = 0; r < 8; ++r) {
          if (mbase + r < M) {
            float v = acc[i][j][r] + bv;
            if (relu) v = fmaxf(v, 0.0f);
            cp[(size_t)r * ldc] = v;
          }
        }
      }
    }
  }
}

// ---------------------------------------------------------------------------
// GAT edge phase
// ---------------------------------------------------------------------------
__global__ void k_att(const float* __restrict__ xp, const float* __restrict__ att_src,
                      const float* __restrict__ att_dst, float* __restrict__ a_src,
                      float* __restrict__ a_dst) {
  int idx = blockIdx.x * blockDim.x + threadIdx.x;
  if (idx >= N_NODES * HEADS) return;
  int n = idx >> 1, h = idx & 1;
  const float* xr = xp + (size_t)n * (HEADS * OUT_C) + h * OUT_C;
  const float* as = att_src + h * OUT_C;
  const float* ad = att_dst + h * OUT_C;
  float s = 0.f, d = 0.f;
  for (int c = 0; c < OUT_C; ++c) { float x = xr[c]; s += x * as[c]; d += x * ad[c]; }
  a_src[idx] = s; a_dst[idx] = d;
}

__global__ void k_init(float* __restrict__ gat, const float* __restrict__ bias_g,
                       float* __restrict__ amax, float* __restrict__ denom) {
  int idx = blockIdx.x * blockDim.x + threadIdx.x;
  if (idx >= N_NODES * 256) return;
  int c = idx & 255, n = idx >> 8;
  gat[idx] = bias_g[c];                  // fold GAT bias into the accumulator
  if (c < 2) { amax[n * 2 + c] = -3.0e38f; denom[n * 2 + c] = 0.f; }
}

__device__ __forceinline__ void atomicMaxF(float* addr, float v) {
  if (v >= 0.f) atomicMax((int*)addr, __float_as_int(v));
  else          atomicMin((unsigned int*)addr, __float_as_uint(v));
}

__device__ __forceinline__ void edgePair(int e, int& src, int& dst, const int* ei) {
  if (e < N_EDGES) { src = ei[e]; dst = ei[N_EDGES + e]; }
  else             { src = dst = e - N_EDGES; }             // self loops
}

__global__ void k_edge_lrelu_max(const int* __restrict__ ei, const float* __restrict__ a_src,
                                 const float* __restrict__ a_dst, float* __restrict__ alpha,
                                 float* __restrict__ amax) {
  int idx = blockIdx.x * blockDim.x + threadIdx.x;
  const int NE = N_EDGES + N_NODES;
  if (idx >= NE * 2) return;
  int e = idx >> 1, h = idx & 1, src, dst;
  edgePair(e, src, dst, ei);
  float v = a_src[src * 2 + h] + a_dst[dst * 2 + h];
  v = v > 0.f ? v : v * NEG_SLOPE;
  alpha[idx] = v;
  atomicMaxF(&amax[dst * 2 + h], v);
}

__global__ void k_edge_exp_sum(const int* __restrict__ ei, float* __restrict__ alpha,
                               const float* __restrict__ amax, float* __restrict__ denom) {
  int idx = blockIdx.x * blockDim.x + threadIdx.x;
  const int NE = N_EDGES + N_NODES;
  if (idx >= NE * 2) return;
  int e = idx >> 1, h = idx & 1, src, dst;
  edgePair(e, src, dst, ei);
  float v = __expf(alpha[idx] - amax[dst * 2 + h]);
  alpha[idx] = v;
  unsafeAtomicAdd(&denom[dst * 2 + h], v);
}

// one wave32 per edge; lane covers 4 ch of head0 + 4 ch of head1 (float4)
__global__ void k_edge_msg(const int* __restrict__ ei, const float* __restrict__ alpha,
                           const float* __restrict__ denom, const float* __restrict__ xp,
                           float* __restrict__ gat) {
  int gid = blockIdx.x * blockDim.x + threadIdx.x;
  int e = gid >> 5, lane = gid & 31;
  const int NE = N_EDGES + N_NODES;
  if (e >= NE) return;
  int src, dst;
  edgePair(e, src, dst, ei);
  float w0 = alpha[e * 2 + 0] / fmaxf(denom[dst * 2 + 0], 1e-16f);
  float w1 = alpha[e * 2 + 1] / fmaxf(denom[dst * 2 + 1], 1e-16f);
  const float4* xs = (const float4*)(xp + (size_t)src * 256);
  float4 v0 = xs[lane];        // channels [lane*4 .. lane*4+3]      (head 0)
  float4 v1 = xs[32 + lane];   // channels [128+lane*4 ..]           (head 1)
  float* gd = gat + (size_t)dst * 256;
  unsafeAtomicAdd(gd + lane * 4 + 0, v0.x * w0);
  unsafeAtomicAdd(gd + lane * 4 + 1, v0.y * w0);
  unsafeAtomicAdd(gd + lane * 4 + 2, v0.z * w0);
  unsafeAtomicAdd(gd + lane * 4 + 3, v0.w * w0);
  unsafeAtomicAdd(gd + 128 + lane * 4 + 0, v1.x * w1);
  unsafeAtomicAdd(gd + 128 + lane * 4 + 1, v1.y * w1);
  unsafeAtomicAdd(gd + 128 + lane * 4 + 2, v1.z * w1);
  unsafeAtomicAdd(gd + 128 + lane * 4 + 3, v1.w * w1);
}

// ---------------------------------------------------------------------------
extern "C" void kernel_launch(void* const* d_in, const int* in_sizes, int n_in,
                              void* d_out, int out_size, void* d_ws, size_t ws_size,
                              hipStream_t stream) {
  const float* z     = (const float*)d_in[0];
  const int*   ei    = (const int*)  d_in[1];
  const float* W1    = (const float*)d_in[2];
  const float* b1    = (const float*)d_in[3];
  const float* W2    = (const float*)d_in[4];
  const float* b2    = (const float*)d_in[5];
  const float* Wg    = (const float*)d_in[6];
  const float* att_s = (const float*)d_in[7];
  const float* att_d = (const float*)d_in[8];
  const float* biasg = (const float*)d_in[9];
  const float* W3    = (const float*)d_in[10];
  const float* b3    = (const float*)d_in[11];
  float* out = (float*)d_out;

  float* ws = (float*)d_ws;
  size_t off = 0;
  float* x1    = ws + off; off += (size_t)N_NODES * 256;
  float* x2    = ws + off; off += (size_t)N_NODES * 512;
  float* xp    = ws + off; off += (size_t)N_NODES * 256;
  float* gat   = ws + off; off += (size_t)N_NODES * 256;
  float* a_src = ws + off; off += (size_t)N_NODES * 2;
  float* a_dst = ws + off; off += (size_t)N_NODES * 2;
  float* amax  = ws + off; off += (size_t)N_NODES * 2;
  float* denom = ws + off; off += (size_t)N_NODES * 2;
  float* alpha = ws + off; off += (size_t)(N_EDGES + N_NODES) * 2;
  unsigned short* W1t = (unsigned short*)(ws + off);   // 16B-aligned (off % 4 == 0)
  unsigned short* W2t = W1t + 512 * 256;
  unsigned short* Wgt = W2t + 256 * 512;
  unsigned short* W3t = Wgt + 512 * 256;

  dim3 blk(256);

  // weight transpose + bf16 convert
  k_wt_bf16<<<(512 * 256 + 255) / 256, blk, 0, stream>>>(W1, W1t, 512, 256);
  k_wt_bf16<<<(256 * 512 + 255) / 256, blk, 0, stream>>>(W2, W2t, 256, 512);
  k_wt_bf16<<<(512 * 256 + 255) / 256, blk, 0, stream>>>(Wg, Wgt, 512, 256);
  k_wt_bf16<<<(256 * 512 + 255) / 256, blk, 0, stream>>>(W3, W3t, 256, 512);

  const int mt = (N_NODES + 63) / 64;   // 313

  // x1 = relu(z @ W1 + b1); x2 = relu(x1 @ W2 + b2); xp = x2 @ Wg
  k_gemm<<<dim3(mt, 2), blk, 0, stream>>>(z,  W1t, b1,      x1, N_NODES, 512, 256, 1);
  k_gemm<<<dim3(mt, 4), blk, 0, stream>>>(x1, W2t, b2,      x2, N_NODES, 256, 512, 1);
  k_gemm<<<dim3(mt, 2), blk, 0, stream>>>(x2, Wgt, nullptr, xp, N_NODES, 512, 256, 0);

  // GAT attention
  k_att<<<(N_NODES * 2 + 255) / 256, blk, 0, stream>>>(xp, att_s, att_d, a_src, a_dst);
  k_init<<<(N_NODES * 256 + 255) / 256, blk, 0, stream>>>(gat, biasg, amax, denom);

  const int NE2 = (N_EDGES + N_NODES) * 2;
  k_edge_lrelu_max<<<(NE2 + 255) / 256, blk, 0, stream>>>(ei, a_src, a_dst, alpha, amax);
  k_edge_exp_sum<<<(NE2 + 255) / 256, blk, 0, stream>>>(ei, alpha, amax, denom);
  k_edge_msg<<<(((size_t)(N_EDGES + N_NODES) * 32 + 255) / 256), blk, 0, stream>>>(
      ei, alpha, denom, xp, gat);

  // out = gat @ W3 + b3   (bias_g already folded into gat accumulator)
  k_gemm<<<dim3(mt, 4), blk, 0, stream>>>(gat, W3t, b3, out, N_NODES, 256, 512, 0);
}
